// LigerDPOLoss_49246095016514
// MI455X (gfx1250) — matrix operand order repaired and checked by reading, not compiled, not run on hardware
//
#include <hip/hip_runtime.h>
#include <hip/hip_bf16.h>
#include <math.h>

// Problem constants (from reference setup_inputs)
#define B2v     4
#define Tv      512
#define Hv      4096
#define Vv      32000
#define Rv      (B2v * Tv)        // 2048 token rows
#define ROWS    16                // token rows per workgroup (one WMMA M tile)
#define NWG     (Rv / ROWS)       // 128 workgroups
#define THREADS 512               // 16 wave32 waves -> 4 waves per SIMD32
#define NWAVES  16
#define KCH     (Hv / 32)         // 128 K-chunks of 32 (bf16 WMMA K)
#define NTILES  (Vv / 16)         // 2000 column tiles of 16 (125 per wave)
#define NSLOT   (NWAVES * 16)     // per-row softmax contributors: 256
#define IGN     (-100)
#define BETA    0.1f

typedef __attribute__((ext_vector_type(16))) __bf16 v16bf;
typedef __attribute__((ext_vector_type(8)))  float  v8f;

union Frag {
  unsigned u[8];
  uint4    q[2];
  v16bf    v;
};

// Pack the two bf16-hi halves of (u0,u1) into one dword: {hi16(u1), hi16(u0)}.
// Single v_perm_b32: tmp = {s0,s1}, selector bytes (MSB..LSB) = 7,6,3,2.
__device__ __forceinline__ unsigned pk_perm(unsigned u1, unsigned u0) {
  return __builtin_amdgcn_perm(u1, u0, 0x07060302u);
}

// Split two fp32 into packed bf16-hi dword + fp32 residuals (exact; the
// residual compiles to a single v_fma_mix_f32_bf16).
__device__ __forceinline__ unsigned split2(float f0, float f1, float& l0, float& l1) {
  unsigned u0 = __float_as_uint(f0), u1 = __float_as_uint(f1);
  l0 = f0 - __uint_as_float(u0 & 0xFFFF0000u);
  l1 = f1 - __uint_as_float(u1 & 0xFFFF0000u);
  return pk_perm(u1, u0);
}

// ---------------------------------------------------------------------------
// Kernel 1: fused x·Wᵀ + online log-softmax + label-logit capture.
// One WG per 16-row token tile; 16 waves sweep interleaved 16-col tiles of V.
// A (x tile) staged once in LDS as bf16 hi/lo planes; W fetched fp32 and
// split on the fly; product via 3 bf16 WMMAs (hi*hi + hi*lo + lo*hi) ≈ fp32.
// Softmax statistics are PER-LANE (no in-loop cross-lane traffic); the 256
// contributors per row are merged once at the end through LDS.
// ---------------------------------------------------------------------------
__global__ void __launch_bounds__(THREADS)
dpo_logp_kernel(const float* __restrict__ x, const float* __restrict__ W,
                const int* __restrict__ target, float* __restrict__ per_tok) {
  extern __shared__ unsigned smem[];
  unsigned* Ahi = smem;                          // [ROWS][Hv/2] packed bf16 pairs
  unsigned* Alo = smem + ROWS * (Hv / 2);
  float*    s_m = (float*)(smem + 2 * ROWS * (Hv / 2));   // [ROWS][NSLOT]
  float*    s_s = s_m + ROWS * NSLOT;                     // [ROWS][NSLOT]
  float*    s_v = s_s + ROWS * NSLOT;                     // [ROWS][NSLOT]

  __shared__ int s_label[ROWS];

  const int tid = threadIdx.x;
  const int wg  = blockIdx.x;

  if (tid < ROWS) s_label[tid] = target[wg * ROWS + tid];

  // Stage the 16 x-rows into LDS as packed bf16 hi/lo planes.
  const float2* x2 = (const float2*)x + (size_t)wg * ROWS * (Hv / 2);
  for (int i = tid; i < ROWS * (Hv / 2); i += THREADS) {
    float2 f = x2[i];
    float l0, l1;
    unsigned hp = split2(f.x, f.y, l0, l1);
    Ahi[i] = hp;
    Alo[i] = pk_perm(__float_as_uint(l1), __float_as_uint(l0));
  }
  __syncthreads();

  const int wave = tid >> 5;
  const int lane = tid & 31;
  const int h    = lane >> 4;   // half-wave: selects K sub-range (A) / K half (B)
  const int rl   = lane & 15;   // A: row within tile; B/D: column within tile

  // Labels for the 8 rows this lane's accumulator covers (M = r + 8*h).
  int lab_r[8];
  #pragma unroll
  for (int r = 0; r < 8; ++r) lab_r[r] = s_label[h * 8 + r];

  float M[8], S[8], Vl[8];
  #pragma unroll
  for (int r = 0; r < 8; ++r) { M[r] = -INFINITY; S[r] = 0.0f; Vl[r] = 0.0f; }

  // A fragment base (dwords): row rl, k-groups [kb + h*8, +8) and [kb+16+h*8, +8)
  const unsigned* Abase_h = Ahi + rl * (Hv / 2) + h * 4;
  const unsigned* Abase_l = Alo + rl * (Hv / 2) + h * 4;

  for (int tile = wave; tile < NTILES; tile += NWAVES) {
    const int col = tile * 16 + rl;
    // B fragment: fp32 of W[col][kb + h*8, +8) and W[col][kb+16+h*8, +8)
    const float4* wp = (const float4*)(W + (size_t)col * Hv) + h * 2;

    v8f acc = {0.0f, 0.0f, 0.0f, 0.0f, 0.0f, 0.0f, 0.0f, 0.0f};

    #pragma unroll 2
    for (int kc = 0; kc < KCH; ++kc) {
      // ---- B: 64 B/lane fp32 from W (4x global_load_b128) ----
      float4 b0 = wp[kc * 8 + 0];
      float4 b1 = wp[kc * 8 + 1];
      float4 b2 = wp[kc * 8 + 4];
      float4 b3 = wp[kc * 8 + 5];

      // ---- A: two ds_load_b128 per plane, matching 16-bit A 16x32 layout ----
      const uint4* ah4 = (const uint4*)(Abase_h + 16 * kc);
      const uint4* al4 = (const uint4*)(Abase_l + 16 * kc);
      Frag ahi, alo;
      ahi.q[0] = ah4[0]; ahi.q[1] = ah4[2];
      alo.q[0] = al4[0]; alo.q[1] = al4[2];

      // Split fp32 -> bf16 hi (perm-packed) + residual lo (perm-packed).
      Frag bhi, blo;
      float l0, l1, l2, l3;
      bhi.u[0] = split2(b0.x, b0.y, l0, l1);
      bhi.u[1] = split2(b0.z, b0.w, l2, l3);
      blo.u[0] = pk_perm(__float_as_uint(l1), __float_as_uint(l0));
      blo.u[1] = pk_perm(__float_as_uint(l3), __float_as_uint(l2));
      bhi.u[2] = split2(b1.x, b1.y, l0, l1);
      bhi.u[3] = split2(b1.z, b1.w, l2, l3);
      blo.u[2] = pk_perm(__float_as_uint(l1), __float_as_uint(l0));
      blo.u[3] = pk_perm(__float_as_uint(l3), __float_as_uint(l2));
      bhi.u[4] = split2(b2.x, b2.y, l0, l1);
      bhi.u[5] = split2(b2.z, b2.w, l2, l3);
      blo.u[4] = pk_perm(__float_as_uint(l1), __float_as_uint(l0));
      blo.u[5] = pk_perm(__float_as_uint(l3), __float_as_uint(l2));
      bhi.u[6] = split2(b3.x, b3.y, l0, l1);
      bhi.u[7] = split2(b3.z, b3.w, l2, l3);
      blo.u[6] = pk_perm(__float_as_uint(l1), __float_as_uint(l0));
      blo.u[7] = pk_perm(__float_as_uint(l3), __float_as_uint(l2));

      // Prefetch next K-chunk of W (speculative; dropped if OOB)
      __builtin_prefetch((const void*)(wp + 8 * (kc + 1)), 0, 1);

      // hi*hi + hi*lo + lo*hi : ~fp32-grade product, fp32 accumulate
      acc = __builtin_amdgcn_wmma_f32_16x16x32_bf16(false, ahi.v, false, bhi.v,
                                                    (short)0, acc, false, false);
      acc = __builtin_amdgcn_wmma_f32_16x16x32_bf16(false, ahi.v, false, blo.v,
                                                    (short)0, acc, false, false);
      acc = __builtin_amdgcn_wmma_f32_16x16x32_bf16(false, alo.v, false, bhi.v,
                                                    (short)0, acc, false, false);
    }

    // ---- Per-lane online softmax update (pure VALU/TRANS, no LDS) ----
    // D layout: lane l, VGPR r -> M = r + 8*(l>>4), N = l&15.
    #pragma unroll
    for (int r = 0; r < 8; ++r) {
      float d = acc[r];
      Vl[r] = (col == lab_r[r]) ? d : Vl[r];      // v_cmp + v_cndmask
      float nm = fmaxf(M[r], d);
      S[r] = S[r] * __expf(M[r] - nm) + __expf(d - nm);
      M[r] = nm;
    }
  }

  // ---- One-time merge of the 256 per-row contributors through LDS ----
  const int slot = wave * 16 + rl;
  #pragma unroll
  for (int r = 0; r < 8; ++r) {
    int row = h * 8 + r;
    s_m[row * NSLOT + slot] = M[r];
    s_s[row * NSLOT + slot] = S[r];
    s_v[row * NSLOT + slot] = Vl[r];
  }
  __syncthreads();

  if (tid < ROWS) {
    const int row = tid;
    float gm = -INFINITY;
    for (int i = 0; i < NSLOT; ++i) gm = fmaxf(gm, s_m[row * NSLOT + i]);
    float gs = 0.0f, lv = 0.0f;
    for (int i = 0; i < NSLOT; ++i) {
      gs += s_s[row * NSLOT + i] * __expf(s_m[row * NSLOT + i] - gm);
      lv += s_v[row * NSLOT + i];   // exactly one contributor captured the label
    }
    int   lb = s_label[row];
    float lp = (lb == IGN) ? 0.0f : (lv - gm - __logf(gs));
    per_tok[wg * ROWS + row] = lp;  // already mask-multiplied (0 when ignored)
  }
}

// ---------------------------------------------------------------------------
// Kernel 2: per-sequence averages -> pref_loss + nll scalar.
// ---------------------------------------------------------------------------
__global__ void __launch_bounds__(256)
dpo_reduce_kernel(const float* __restrict__ per_tok,
                  const int* __restrict__ target,
                  float* __restrict__ out) {
  __shared__ float ssum[B2v];
  __shared__ int   scnt[B2v];
  const int tid = threadIdx.x;
  if (tid < B2v) { ssum[tid] = 0.0f; scnt[tid] = 0; }
  __syncthreads();

  float part[B2v]  = {0.0f, 0.0f, 0.0f, 0.0f};
  int   cpart[B2v] = {0, 0, 0, 0};
  for (int i = tid; i < Rv; i += 256) {
    int s = i / Tv;
    part[s]  += per_tok[i];
    cpart[s] += (target[i] != IGN) ? 1 : 0;
  }
  #pragma unroll
  for (int s = 0; s < B2v; ++s) {
    atomicAdd(&ssum[s], part[s]);
    atomicAdd(&scnt[s], cpart[s]);
  }
  __syncthreads();

  if (tid == 0) {
    float avg[B2v];
    #pragma unroll
    for (int s = 0; s < B2v; ++s) avg[s] = ssum[s] / (float)scnt[s];
    const int Bh = B2v / 2;
    float pref = 0.0f;
    for (int b = 0; b < Bh; ++b) {
      // ref_* == stop_gradient(*) equals the value in forward -> diff is 0,
      // but compute it literally per the reference.
      float diff = BETA * ((avg[b] - avg[b]) - (avg[Bh + b] - avg[Bh + b]));
      pref += __logf(1.0f + __expf(-diff));   // -log_sigmoid(diff)
    }
    pref /= (float)Bh;
    float nll = -(ssum[0] + ssum[1]) / (float)(scnt[0] + scnt[1]);
    out[0] = pref + nll;
  }
}

extern "C" void kernel_launch(void* const* d_in, const int* in_sizes, int n_in,
                              void* d_out, int out_size, void* d_ws, size_t ws_size,
                              hipStream_t stream) {
  const float* x      = (const float*)d_in[0];
  const int*   target = (const int*)d_in[1];
  const float* W      = (const float*)d_in[2];
  float*       out    = (float*)d_out;
  float*       ptok   = (float*)d_ws;          // 2048 floats = 8 KB

  // A hi/lo planes (256 KB) + merge scratch (48 KB) = 304 KB (< 320 KB/WG)
  const size_t smem = (size_t)2 * ROWS * (Hv / 2) * sizeof(unsigned)
                    + (size_t)3 * ROWS * NSLOT * sizeof(float);
  hipFuncSetAttribute((const void*)dpo_logp_kernel,
                      hipFuncAttributeMaxDynamicSharedMemorySize, (int)smem);

  dpo_logp_kernel<<<NWG, THREADS, smem, stream>>>(x, W, target, ptok);
  dpo_reduce_kernel<<<1, 256, 0, stream>>>(ptok, target, out);
}
